// ComENet_46394236731778
// MI455X (gfx1250) — matrix operand order repaired and checked by reading, not compiled, not run on hardware
//
#include <hip/hip_runtime.h>
#include <hip/hip_bf16.h>
#include <cstdint>

// ---------------------------------------------------------------------------
// ComENet forward for MI455X (gfx1250, wave32, WMMA).
//
// - Edge MLP folded algebraically: Wc = W2@W1 (256x12), so the giant
//   E x 256 x 256 GEMMs disappear; edge projection + gather + scatter-add is
//   one fused atomic kernel.
// - All node-level H x H GEMMs -> bf16 WMMA (v_wmma_f32_16x16x32_bf16,
//   f32 accumulate), weights staged in LDS via global_load_async_to_lds_b128
//   (ASYNCcnt path), fragments read back with ds_load_b128.
// - DUAL is a template parameter so the dual-accumulate GEMMs
//   (a1@c1_rel^T + xi@c1_root^T, and the split-K lincat) are branch-free.
// ---------------------------------------------------------------------------

typedef __attribute__((ext_vector_type(16))) __bf16 v16bf;
typedef __attribute__((ext_vector_type(8)))  float  v8f;

__device__ __forceinline__ float swishf(float v) {
  return v / (1.0f + __expf(-v));
}

// Async-copy one 16B chunk from global to LDS (CDNA5 async path, ASYNCcnt).
__device__ __forceinline__ void async_g2l_b128(uint32_t lds_off,
                                               const void* gptr) {
  uint64_t ga = (uint64_t)(uintptr_t)gptr;
  asm volatile("global_load_async_to_lds_b128 %0, %1, off"
               :: "v"(lds_off), "v"(ga)
               : "memory");
}

__device__ __forceinline__ void wait_async_all() {
  asm volatile("s_wait_asynccnt 0x0" ::: "memory");
}

// ---------------------------------------------------------------------------
// Fused GEMM:  Y[M,Nout] = post( A[M,256] @ W[Nout, ldW]^T
//                                (+ A2 @ W2^T) + bias )  (+ res)
// post = swish if act==1, applied BEFORE residual add (matches reference).
// K fixed at 256. Nout multiple of 64. Tile: 128 rows x 64 cols, 256 threads
// = 8 waves; each wave owns 16 rows x 64 cols (4 WMMA accumulators).
// ---------------------------------------------------------------------------
template <bool DUAL>
__global__ __launch_bounds__(256) void gemm_wmma_k(
    const float* __restrict__ A,
    const __bf16* __restrict__ W, int ldW,
    const float* __restrict__ A2,
    const __bf16* __restrict__ W2, int ldW2,
    const float* __restrict__ bias,
    const float* __restrict__ res,
    float* __restrict__ Y,
    int Mrows, int Nout, int act)
{
  constexpr int K = 256;
  // 64 rows(cols of Y) x 256 K of bf16 = 32KB per matrix.
  __shared__ __align__(32) __bf16 sW[(DUAL ? 2 : 1) * 64 * K];

  const int tid     = threadIdx.x;
  const int colBase = blockIdx.y * 64;

  // ---- Stage weight tile(s) into LDS with async 16B copies. ----
  // 64 rows * 512B/row = 2048 chunks of 16B; 8 chunks per thread.
  {
    const uint32_t ldsBase = (uint32_t)(uintptr_t)&sW[0];
    #pragma unroll
    for (int it = 0; it < 8; ++it) {
      int c = tid + it * 256;
      int row = c >> 5;              // 32 chunks per 512B row
      int elt = (c & 31) * 8;        // bf16 elements within row
      async_g2l_b128(ldsBase + (uint32_t)c * 16,
                     W + (size_t)(colBase + row) * ldW + elt);
    }
    if (DUAL) {
      #pragma unroll
      for (int it = 0; it < 8; ++it) {
        int c = tid + it * 256;
        int row = c >> 5;
        int elt = (c & 31) * 8;
        async_g2l_b128(ldsBase + (uint32_t)(64 * K * 2) + (uint32_t)c * 16,
                       W2 + (size_t)(colBase + row) * ldW2 + elt);
      }
    }
    wait_async_all();
  }
  __syncthreads();

  const int wave    = tid >> 5;
  const int lane    = tid & 31;
  const int hiHalf  = lane >> 4;   // A: K 0..15 vs 16..31
  const int lrow    = lane & 15;
  const int rowBase = blockIdx.x * 128 + wave * 16;

  v8f acc[4] = {};

  int rowA = rowBase + lrow;
  if (rowA >= Mrows) rowA = Mrows - 1;     // clamp loads; stores are guarded
  const float* aRow  = A + (size_t)rowA * K;
  const float* a2Row = DUAL ? (A2 + (size_t)rowA * K) : nullptr;

  const __bf16* sW2p = sW + 64 * K;        // valid only when DUAL

  #pragma unroll
  for (int kb = 0; kb < K / 32; ++kb) {
    // A fragment: 16x32 bf16; lane holds row lrow, two 8-wide K chunks.
    const int k0 = kb * 32 + hiHalf * 16;
    v16bf afrag;
    #pragma unroll
    for (int j = 0; j < 8; ++j) {
      afrag[j]     = (__bf16)aRow[k0 + j];
      afrag[8 + j] = (__bf16)aRow[k0 + 8 + j];
    }
    v16bf a2frag;
    if (DUAL) {
      #pragma unroll
      for (int j = 0; j < 8; ++j) {
        a2frag[j]     = (__bf16)a2Row[k0 + j];
        a2frag[8 + j] = (__bf16)a2Row[k0 + 8 + j];
      }
    }
    // Load all B fragments first so ds_loads overlap with WMMA issue.
    const int kB = kb * 32 + hiHalf * 16;
    v16bf bf[4];
    #pragma unroll
    for (int c = 0; c < 4; ++c)
      bf[c] = *(const v16bf*)&sW[(c * 16 + lrow) * K + kB];
    v16bf b2[4];
    if (DUAL) {
      #pragma unroll
      for (int c = 0; c < 4; ++c)
        b2[c] = *(const v16bf*)&sW2p[(c * 16 + lrow) * K + kB];
    }
    #pragma unroll
    for (int c = 0; c < 4; ++c) {
      acc[c] = __builtin_amdgcn_wmma_f32_16x16x32_bf16(
          false, afrag, false, bf[c], (short)0, acc[c], false, false);
      if (DUAL) {
        acc[c] = __builtin_amdgcn_wmma_f32_16x16x32_bf16(
            false, a2frag, false, b2[c], (short)0, acc[c], false, false);
      }
    }
  }

  // D layout: VGPR i -> (M = i + 8*hiHalf, N = lrow) within each 16x16 tile.
  const int rOff = hiHalf * 8;
  #pragma unroll
  for (int c = 0; c < 4; ++c) {
    const int col = colBase + c * 16 + lrow;
    const float bv = bias ? bias[col] : 0.0f;
    #pragma unroll
    for (int i = 0; i < 8; ++i) {
      int r = rowBase + rOff + i;
      if (r < Mrows) {
        float v = acc[c][i] + bv;
        if (act) v = swishf(v);
        if (res) v += res[(size_t)r * Nout + col];
        Y[(size_t)r * Nout + col] = v;
      }
    }
  }
}

// ---------------------------------------------------------------------------
// Small helper kernels
// ---------------------------------------------------------------------------
__global__ void zero_k(float* p, size_t n) {
  size_t i = (size_t)blockIdx.x * blockDim.x + threadIdx.x;
  size_t stride = (size_t)gridDim.x * blockDim.x;
  for (; i < n; i += stride) p[i] = 0.0f;
}

__global__ void cvt_bf16_k(const float* __restrict__ s, __bf16* __restrict__ d,
                           size_t n) {
  size_t i = (size_t)blockIdx.x * blockDim.x + threadIdx.x;
  size_t stride = (size_t)gridDim.x * blockDim.x;
  for (; i < n; i += stride) d[i] = (__bf16)s[i];
}

// x[n,h] = swish(emb[z[n], h])
__global__ void node_init_k(const float* __restrict__ emb,
                            const int* __restrict__ z,
                            float* __restrict__ x, int N) {
  int n = blockIdx.x;
  if (n >= N) return;
  int h = threadIdx.x;
  float e = emb[(size_t)z[n] * 256 + h];
  x[(size_t)n * 256 + h] = swishf(e);
}

// Spherical-Bessel / spherical-harmonic edge features.
__global__ void edge_feat_k(const float* __restrict__ dist,
                            const float* __restrict__ theta,
                            const float* __restrict__ phi,
                            const float* __restrict__ tau,
                            float* __restrict__ f1, float* __restrict__ f2,
                            int E) {
  int e = blockIdx.x * blockDim.x + threadIdx.x;
  if (e >= E) return;
  const float NPIc[3] = {3.14159265f, 6.28318531f, 9.42477796f};
  const float Z1c[3]  = {4.4934094579f, 7.7252518369f, 10.9041216594f};
  const float N1c[3]  = {6.51012f, 11.01623f, 15.48548f};  // sqrt2/|j2(z1)|
  const float C0 = 0.28209479177f, C1 = 0.48860251190f, SQ2 = 1.41421356f;

  float xd = dist[e] * 0.125f;   // dist / CUTOFF
  float l0[3], l1[3];
  #pragma unroll
  for (int r = 0; r < 3; ++r) {
    l0[r] = SQ2 * __sinf(xd * NPIc[r]) / xd;
    float t = xd * Z1c[r];
    l1[r] = N1c[r] * (__sinf(t) / (t * t) - __cosf(t) / t);
  }
  float st = __sinf(theta[e]), ct = __cosf(theta[e]);
  float sp = __sinf(phi[e]),   cp = __cosf(phi[e]);
  float s4[4] = {C0, C1 * st * sp, C1 * ct, C1 * st * cp};
  float sa[2] = {C0, C1 * __cosf(tau[e])};

  float* o1 = f1 + (size_t)e * 12;
  float* o2 = f2 + (size_t)e * 6;
  #pragma unroll
  for (int a = 0; a < 4; ++a)
    #pragma unroll
    for (int b = 0; b < 3; ++b)
      o1[a * 3 + b] = (a == 0 ? l0[b] : l1[b]) * s4[a];
  #pragma unroll
  for (int a = 0; a < 2; ++a)
    #pragma unroll
    for (int b = 0; b < 3; ++b)
      o2[a * 3 + b] = (a == 0 ? l0[b] : l1[b]) * sa[a];
}

// Wc[h,j] = sum_m W2[h,m] * W1[m,j]   (fold the two edge linears together)
__global__ void combine_w_k(const float* __restrict__ W2,
                            const float* __restrict__ W1,
                            float* __restrict__ out, int J) {
  int idx = blockIdx.x * blockDim.x + threadIdx.x;
  if (idx >= 256 * J) return;
  int h = idx / J, j = idx % J;
  float s = 0.0f;
  for (int m = 0; m < 256; ++m) s += W2[h * 256 + m] * W1[m * J + j];
  out[idx] = s;
}

__global__ void count_nodes_k(const int* __restrict__ batch,
                              float* __restrict__ cnt, int N) {
  int n = blockIdx.x * blockDim.x + threadIdx.x;
  if (n < N) atomicAdd(&cnt[batch[n]], 1.0f);
}

// Fused edge message: a1[dst] += (feat1@Wc1^T)*xi[src]; same for a2/Wc2.
// One wave per edge; lane covers 8 of the 256 channels.
__global__ __launch_bounds__(256) void edge_scatter_k(
    const float* __restrict__ feat1, const float* __restrict__ feat2,
    const float* __restrict__ wc1, const float* __restrict__ wc2,
    const float* __restrict__ xi, const int* __restrict__ src,
    const int* __restrict__ dst, float* __restrict__ a1,
    float* __restrict__ a2, int E)
{
  __shared__ float sW1[256 * 12];
  __shared__ float sW2[256 * 6];
  int tid = threadIdx.x;
  for (int i = tid; i < 256 * 12; i += 256) sW1[i] = wc1[i];
  for (int i = tid; i < 256 * 6;  i += 256) sW2[i] = wc2[i];
  __syncthreads();

  int e = blockIdx.x * 8 + (tid >> 5);
  if (e >= E) return;
  int lane = tid & 31;
  int s = src[e], d = dst[e];

  float ft1[12], ft2[6];
  #pragma unroll
  for (int j = 0; j < 12; ++j) ft1[j] = feat1[(size_t)e * 12 + j];
  #pragma unroll
  for (int j = 0; j < 6;  ++j) ft2[j] = feat2[(size_t)e * 6 + j];

  const float* xr = xi + (size_t)s * 256;
  float* o1 = a1 + (size_t)d * 256;
  float* o2 = a2 + (size_t)d * 256;

  #pragma unroll
  for (int m = 0; m < 8; ++m) {
    int h = lane + 32 * m;
    const float* w1 = &sW1[h * 12];
    const float* w2 = &sW2[h * 6];
    float v1 = 0.0f, v2 = 0.0f;
    #pragma unroll
    for (int j = 0; j < 12; ++j) v1 += w1[j] * ft1[j];
    #pragma unroll
    for (int j = 0; j < 6;  ++j) v2 += w2[j] * ft2[j];
    float xv = xr[h];
    atomicAdd(o1 + h, v1 * xv);
    atomicAdd(o2 + h, v2 * xv);
  }
}

// Graph-norm kernels
__global__ void accum_sum_k(const float* __restrict__ h,
                            const int* __restrict__ batch,
                            float* __restrict__ msum, int N) {
  int n = blockIdx.x;
  if (n >= N) return;
  int c = threadIdx.x;
  atomicAdd(&msum[(size_t)batch[n] * 256 + c], h[(size_t)n * 256 + c]);
}

__global__ void center_var_k(const float* __restrict__ h,
                             const int* __restrict__ batch,
                             const float* __restrict__ msum,
                             const float* __restrict__ cnt,
                             const float* __restrict__ ms,
                             float* __restrict__ hc,
                             float* __restrict__ vsum, int N) {
  int n = blockIdx.x;
  if (n >= N) return;
  int c = threadIdx.x;
  int g = batch[n];
  float cc = fmaxf(cnt[g], 1.0f);
  float mean = msum[(size_t)g * 256 + c] / cc;
  float v = h[(size_t)n * 256 + c] - mean * ms[c];
  hc[(size_t)n * 256 + c] = v;
  atomicAdd(&vsum[(size_t)g * 256 + c], v * v);
}

__global__ void normalize_k(const float* __restrict__ hc,
                            const int* __restrict__ batch,
                            const float* __restrict__ vsum,
                            const float* __restrict__ cnt,
                            const float* __restrict__ w,
                            const float* __restrict__ b,
                            float* __restrict__ out, int N) {
  int n = blockIdx.x;
  if (n >= N) return;
  int c = threadIdx.x;
  int g = batch[n];
  float cc = fmaxf(cnt[g], 1.0f);
  float var = vsum[(size_t)g * 256 + c] / cc;
  out[(size_t)n * 256 + c] =
      w[c] * hc[(size_t)n * 256 + c] * __frsqrt_rn(var + 1e-5f) + b[c];
}

// Final pooling: out[g, 0:128] += xo[n, 0:128]
__global__ void segsum_out_k(const float* __restrict__ xo,
                             const int* __restrict__ batch,
                             float* __restrict__ out, int N) {
  int n = blockIdx.x;
  if (n >= N) return;
  int c = threadIdx.x;
  atomicAdd(&out[(size_t)batch[n] * 128 + c], xo[(size_t)n * 128 + c]);
}

// ---------------------------------------------------------------------------
// Host launcher
// ---------------------------------------------------------------------------
extern "C" void kernel_launch(void* const* d_in, const int* in_sizes, int n_in,
                              void* d_out, int out_size, void* d_ws,
                              size_t ws_size, hipStream_t stream) {
  // Input order: jax pytree flatten of setup_inputs() dict:
  // params leaves (dicts sorted by key, recursively), then z, dist, theta,
  // phi, tau, edge_index, batch.
  const float* p_c1_rel_W  = (const float*)d_in[0];
  const float* p_c1_rel_b  = (const float*)d_in[1];
  const float* p_c1_root_W = (const float*)d_in[2];
  const float* p_c2_rel_W  = (const float*)d_in[3];
  const float* p_c2_rel_b  = (const float*)d_in[4];
  const float* p_c2_root_W = (const float*)d_in[5];
  const float* p_f1_W1     = (const float*)d_in[6];
  const float* p_f1_W2     = (const float*)d_in[7];
  const float* p_f2_W1     = (const float*)d_in[8];
  const float* p_f2_W2     = (const float*)d_in[9];
  const float* p_final_W   = (const float*)d_in[10];
  const float* p_final_b   = (const float*)d_in[11];
  const float* p_lin1_W    = (const float*)d_in[12];
  const float* p_lin1_b    = (const float*)d_in[13];
  const float* p_lin2_W    = (const float*)d_in[14];
  const float* p_lin2_b    = (const float*)d_in[15];
  const float* p_lin_W     = (const float*)d_in[16];
  const float* p_lin_b     = (const float*)d_in[17];
  const float* p_lincat_W  = (const float*)d_in[18];
  const float* p_lincat_b  = (const float*)d_in[19];
  const float* p_lins_W    = (const float*)d_in[20];
  const float* p_lins_b    = (const float*)d_in[21];
  const float* p_norm_b    = (const float*)d_in[22];
  const float* p_norm_ms   = (const float*)d_in[23];
  const float* p_norm_w    = (const float*)d_in[24];
  const float* p_emb       = (const float*)d_in[25];
  const float* p_linout_W  = (const float*)d_in[26];
  const float* p_linout_b  = (const float*)d_in[27];
  const float* p_outlins_W = (const float*)d_in[28];
  const float* p_outlins_b = (const float*)d_in[29];
  const int*   z_in        = (const int*)d_in[30];
  const float* dist        = (const float*)d_in[31];
  const float* theta       = (const float*)d_in[32];
  const float* phi         = (const float*)d_in[33];
  const float* tau         = (const float*)d_in[34];
  const int*   ei          = (const int*)d_in[35];
  const int*   batch       = (const int*)d_in[36];

  const int N = in_sizes[30];
  const int E = in_sizes[31];
  const int* e_src = ei;
  const int* e_dst = ei + E;

  const size_t NH = (size_t)N * 256;

  // --- workspace carve ---
  float* wsf = (float*)d_ws;
  size_t off = 0;
  auto alloc = [&](size_t n) { float* p = wsf + off; off += n; return p; };
  float* X   = alloc(NH);
  float* XI  = alloc(NH);
  float* A1  = alloc(NH);
  float* A2  = alloc(NH);
  float* B1  = alloc(NH);
  float* B2  = alloc(NH);
  float* B3  = alloc(NH);
  float* F1  = alloc((size_t)E * 12);
  float* F2  = alloc((size_t)E * 6);
  float* XO  = alloc((size_t)N * 128);
  float* CNT = alloc(512);
  float* MS  = alloc(512 * 256);
  float* VS  = alloc(512 * 256);
  float* WC1 = alloc(4 * 256 * 12);
  float* WC2 = alloc(4 * 256 * 6);
  __bf16* wb = (__bf16*)(wsf + off);

  const size_t HH = 65536;  // 256*256
  __bf16* wbLin    = wb;                 // 4*HH
  __bf16* wbC1rel  = wbLin    + 4 * HH;
  __bf16* wbC1root = wbC1rel  + 4 * HH;
  __bf16* wbC2rel  = wbC1root + 4 * HH;
  __bf16* wbC2root = wbC2rel  + 4 * HH;
  __bf16* wbLin1   = wbC2root + 4 * HH;
  __bf16* wbLin2   = wbLin1   + 4 * HH;
  __bf16* wbLincat = wbLin2   + 4 * HH;       // 4 * 131072
  __bf16* wbLins   = wbLincat + 4 * 131072;   // 4*3*HH
  __bf16* wbFinal  = wbLins   + 12 * HH;
  __bf16* wbOutl   = wbFinal  + 4 * HH;       // 3*HH
  __bf16* wbLinout = wbOutl   + 3 * HH;       // 128*256

  auto zero = [&](float* p, size_t n) {
    int g = (int)((n + 255) / 256); if (g > 4096) g = 4096;
    zero_k<<<dim3(g), dim3(256), 0, stream>>>(p, n);
  };
  auto cvt = [&](const float* s, __bf16* d, size_t n) {
    int g = (int)((n + 255) / 256); if (g > 4096) g = 4096;
    cvt_bf16_k<<<dim3(g), dim3(256), 0, stream>>>(s, d, n);
  };
  auto gemm1 = [&](const float* A, const __bf16* W, int ldW,
                   const float* bias, const float* res, float* Y, int Mr,
                   int Nout, int act) {
    dim3 g((Mr + 127) / 128, Nout / 64);
    gemm_wmma_k<false><<<g, dim3(256), 0, stream>>>(
        A, W, ldW, nullptr, nullptr, 0, bias, res, Y, Mr, Nout, act);
  };
  auto gemm2 = [&](const float* A, const __bf16* W, int ldW, const float* Ab,
                   const __bf16* Wb, int ldWb, const float* bias,
                   const float* res, float* Y, int Mr, int Nout, int act) {
    dim3 g((Mr + 127) / 128, Nout / 64);
    gemm_wmma_k<true><<<g, dim3(256), 0, stream>>>(
        A, W, ldW, Ab, Wb, ldWb, bias, res, Y, Mr, Nout, act);
  };

  // --- prep ---
  zero(CNT, 512);
  count_nodes_k<<<dim3((N + 255) / 256), dim3(256), 0, stream>>>(batch, CNT, N);
  node_init_k<<<dim3(N), dim3(256), 0, stream>>>(p_emb, z_in, X, N);
  edge_feat_k<<<dim3((E + 255) / 256), dim3(256), 0, stream>>>(
      dist, theta, phi, tau, F1, F2, E);

  cvt(p_lin_W,     wbLin,    4 * HH);
  cvt(p_c1_rel_W,  wbC1rel,  4 * HH);
  cvt(p_c1_root_W, wbC1root, 4 * HH);
  cvt(p_c2_rel_W,  wbC2rel,  4 * HH);
  cvt(p_c2_root_W, wbC2root, 4 * HH);
  cvt(p_lin1_W,    wbLin1,   4 * HH);
  cvt(p_lin2_W,    wbLin2,   4 * HH);
  cvt(p_lincat_W,  wbLincat, 4 * 131072);
  cvt(p_lins_W,    wbLins,   12 * HH);
  cvt(p_final_W,   wbFinal,  4 * HH);
  cvt(p_outlins_W, wbOutl,   3 * HH);
  cvt(p_linout_W,  wbLinout, 128 * 256);

  for (int i = 0; i < 4; ++i) {
    combine_w_k<<<dim3((256 * 12 + 255) / 256), dim3(256), 0, stream>>>(
        p_f1_W2 + (size_t)i * HH, p_f1_W1 + (size_t)i * 256 * 12,
        WC1 + (size_t)i * 256 * 12, 12);
    combine_w_k<<<dim3((256 * 6 + 255) / 256), dim3(256), 0, stream>>>(
        p_f2_W2 + (size_t)i * HH, p_f2_W1 + (size_t)i * 256 * 6,
        WC2 + (size_t)i * 256 * 6, 6);
  }

  // --- interaction blocks ---
  for (int i = 0; i < 4; ++i) {
    const size_t iH = (size_t)i * 256;
    // xi = swish(x @ lin_W^T + lin_b)
    gemm1(X, wbLin + i * HH, 256, p_lin_b + iH, nullptr, XI, N, 256, 1);
    // edge messages
    zero(A1, NH);
    zero(A2, NH);
    edge_scatter_k<<<dim3((E + 7) / 8), dim3(256), 0, stream>>>(
        F1, F2, WC1 + (size_t)i * 256 * 12, WC2 + (size_t)i * 256 * 6, XI,
        e_src, e_dst, A1, A2, E);
    // h1 = swish((a1@c1_rel^T + b + xi@c1_root^T) @ lin1^T + lin1_b)
    gemm2(A1, wbC1rel + i * HH, 256, XI, wbC1root + i * HH, 256,
          p_c1_rel_b + iH, nullptr, B1, N, 256, 0);
    gemm1(B1, wbLin1 + i * HH, 256, p_lin1_b + iH, nullptr, B2, N, 256, 1);
    // h2
    gemm2(A2, wbC2rel + i * HH, 256, XI, wbC2root + i * HH, 256,
          p_c2_rel_b + iH, nullptr, B1, N, 256, 0);
    gemm1(B1, wbLin2 + i * HH, 256, p_lin2_b + iH, nullptr, B3, N, 256, 1);
    // h = [h1|h2] @ lincat^T + b + xi   (split K=512 into dual K=256)
    gemm2(B2, wbLincat + (size_t)i * 131072, 512, B3,
          wbLincat + (size_t)i * 131072 + 256, 512, p_lincat_b + iH, XI, B1,
          N, 256, 0);
    // 3 residual MLP layers: h = swish(h@W+b) + h   (ping-pong B1<->B2)
    float* hin = B1; float* hout = B2;
    for (int j = 0; j < 3; ++j) {
      gemm1(hin, wbLins + (size_t)(i * 3 + j) * HH, 256,
            p_lins_b + (size_t)(i * 3 + j) * 256, hin, hout, N, 256, 1);
      float* t = hin; hin = hout; hout = t;
    }
    // graph norm (result of loop is in `hin`)
    zero(MS, 512 * 256);
    zero(VS, 512 * 256);
    accum_sum_k<<<dim3(N), dim3(256), 0, stream>>>(hin, batch, MS, N);
    center_var_k<<<dim3(N), dim3(256), 0, stream>>>(
        hin, batch, MS, CNT, p_norm_ms + iH, B3, VS, N);
    normalize_k<<<dim3(N), dim3(256), 0, stream>>>(
        B3, batch, VS, CNT, p_norm_w + iH, p_norm_b + iH, hout, N);
    // x = hnorm @ final^T + final_b
    gemm1(hout, wbFinal + i * HH, 256, p_final_b + iH, nullptr, X, N, 256, 0);
  }

  // --- output head ---
  gemm1(X,  wbOutl + 0 * HH, 256, p_outlins_b + 0,   nullptr, XI, N, 256, 1);
  gemm1(XI, wbOutl + 1 * HH, 256, p_outlins_b + 256, nullptr, X,  N, 256, 1);
  gemm1(X,  wbOutl + 2 * HH, 256, p_outlins_b + 512, nullptr, XI, N, 256, 1);
  gemm1(XI, wbLinout, 256, p_linout_b, nullptr, XO, N, 128, 0);

  zero((float*)d_out, (size_t)out_size);
  segsum_out_k<<<dim3(N), dim3(128), 0, stream>>>(XO, batch, (float*)d_out, N);
}
